// APM_9242769621730
// MI455X (gfx1250) — matrix operand order repaired and checked by reading, not compiled, not run on hardware
//
#include <hip/hip_runtime.h>

#define B_  4
#define C_  512
#define CO_ 32
#define T_  16
#define HW_ 784
#define TOUT_ 48
#define KVPAD_ 800
// TEMP * log2(e) : exp(TEMP*s) == exp2(s * TEMP_LOG2E)
#define TEMP_LOG2E 5.7707801635558535f

typedef __attribute__((ext_vector_type(8)))  __bf16 v8bf;
typedef __attribute__((ext_vector_type(16))) __bf16 v16bf;
typedef __attribute__((ext_vector_type(8)))  float  v8f;

static __device__ __forceinline__ v16bf cat8(v8bf a, v8bf b) {
  return __builtin_shufflevector(a, b, 0, 1, 2, 3, 4, 5, 6, 7,
                                 8, 9, 10, 11, 12, 13, 14, 15);
}

// ---------------------------------------------------------------------------
// Kernel 0: x (fp32) -> x_bf16, same (b,c,t,hw) layout.  One streaming pass;
// x_bf16 (51 MB) then lives in the 192 MB L2 for the attention GEMM.
// ---------------------------------------------------------------------------
__global__ __launch_bounds__(256) void cvt_kernel(const float* __restrict__ x,
                                                  __bf16* __restrict__ xb) {
  size_t i = ((size_t)blockIdx.x * blockDim.x + threadIdx.x) * 8;
  float4 f0 = *(const float4*)(x + i);
  float4 f1 = *(const float4*)(x + i + 4);
  v8bf o;
  o[0] = (__bf16)f0.x; o[1] = (__bf16)f0.y;
  o[2] = (__bf16)f0.z; o[3] = (__bf16)f0.w;
  o[4] = (__bf16)f1.x; o[5] = (__bf16)f1.y;
  o[6] = (__bf16)f1.z; o[7] = (__bf16)f1.w;
  *(v8bf*)(xb + i) = o;
}

// ---------------------------------------------------------------------------
// Kernel 1: semantic = Ws @ x (32x512 per position), L2-normalize over co,
// store bf16 x_norm TRANSPOSED: (b, t, hw, co) so WMMA A/B fragments are
// contiguous 8-element runs (b128 loads).
// ---------------------------------------------------------------------------
__global__ __launch_bounds__(256) void semnorm_kernel(
    const float* __restrict__ x, const float* __restrict__ Ws,
    __bf16* __restrict__ xn) {
  int bt = blockIdx.x;            // b*T + t
  int t = bt % T_, b = bt / T_;
  const float* xb = x + ((size_t)b * C_) * (T_ * HW_) + (size_t)t * HW_;
  __bf16* xo = xn + (size_t)bt * (HW_ * CO_);

  for (int p = threadIdx.x; p < HW_; p += blockDim.x) {
    float acc[CO_];
#pragma unroll
    for (int o = 0; o < CO_; ++o) acc[o] = 0.f;
    for (int c = 0; c < C_; ++c) {
      float v = xb[(size_t)c * (T_ * HW_) + p];
#pragma unroll
      for (int o = 0; o < CO_; ++o) acc[o] += Ws[o * C_ + c] * v;
    }
    float ss = 0.f;
#pragma unroll
    for (int o = 0; o < CO_; ++o) ss += acc[o] * acc[o];
    float inv = 1.0f / fmaxf(sqrtf(ss), 1e-12f);
    v8bf pk[4];
#pragma unroll
    for (int o = 0; o < CO_; ++o) pk[o >> 3][o & 7] = (__bf16)(acc[o] * inv);
    __bf16* row = xo + (size_t)p * CO_;
    *(v8bf*)(row)      = pk[0];
    *(v8bf*)(row + 8)  = pk[1];
    *(v8bf*)(row + 16) = pk[2];
    *(v8bf*)(row + 24) = pk[3];
  }
}

// ---------------------------------------------------------------------------
// Kernel 2: one wave32 per (b, t, n, 16-row query tile).
//   Phase 1: S = q.kT (K=co=32), P = exp2(S*TEMP*log2e) -> LDS (kv pad 800),
//            rowsums via half-wave shuffles.  |S|<=TEMP -> no max needed,
//            and raw v_exp_f32 (__builtin_amdgcn_exp2f) is safe.
//   Phase 2: O = (P/rs) . V, channel loop blocked 8-wide so one LDS P
//            fragment feeds 8 WMMAs; V streamed as b128 bf16.
// ---------------------------------------------------------------------------
__global__ __launch_bounds__(32) void attn_kernel(
    const __bf16* __restrict__ xbf, const __bf16* __restrict__ xn,
    float* __restrict__ out) {
  int blk  = blockIdx.x;
  int tile = blk % 49;
  int n    = (blk / 49) & 1;
  int t    = (blk / 98) % T_;
  int b    = blk / (98 * T_);
  int lane = threadIdx.x;

  int  m0   = tile * 16;
  int  Mloc = lane & 15;
  bool hi   = lane >= 16;
  int  tout = t * 3 + (n == 0 ? 0 : 2);
  float* obase = out + ((size_t)b * C_) * (TOUT_ * HW_) + (size_t)tout * HW_
               + m0 + (hi ? 8 : 0);

  int tsrc = t + (n == 0 ? -1 : 1);
  if (tsrc < 0 || tsrc >= T_) {
    // zero-padded neighbor frame -> O == 0 exactly; just write zeros.
    float4 z = {0.f, 0.f, 0.f, 0.f};
    for (int cc = 0; cc < 32; ++cc) {
      float* optr = obase + (size_t)(cc * 16 + Mloc) * (TOUT_ * HW_);
      *(float4*)(optr)     = z;
      *(float4*)(optr + 4) = z;
    }
    return;
  }

  __shared__ __align__(16) __bf16 Plds[16 * KVPAD_];
  __shared__ float rs_lds[16];

  // zero the kv padding region [784, 800)
  if (lane < 16) {
#pragma unroll
    for (int m = 0; m < 16; ++m)
      Plds[m * KVPAD_ + 784 + lane] = (__bf16)0.0f;
  }

  // x_norm transposed layout: row(p) = xn[(bt*HW + p)*CO + ch]
  const __bf16* qT = xn + ((size_t)(b * T_ + t)) * (HW_ * CO_);
  const __bf16* kT = xn + ((size_t)(b * T_ + tsrc)) * (HW_ * CO_);
  // V bf16: [ch][kv] within frame tsrc
  const __bf16* vfrm = xbf + ((size_t)b * C_) * (T_ * HW_) + (size_t)tsrc * HW_;

  // A fragment (q): rows m0..m0+15, K = 32 channels; two contiguous runs
  const __bf16* qrow = qT + (size_t)(m0 + Mloc) * CO_ + (hi ? 16 : 0);
  v16bf afrag = cat8(*(const v8bf*)(qrow), *(const v8bf*)(qrow + 8));

  float rsum[8];
#pragma unroll
  for (int j = 0; j < 8; ++j) rsum[j] = 0.f;

  // ---- phase 1 -----------------------------------------------------------
  for (int kc = 0; kc < 49; ++kc) {
    int kv0 = kc * 16;
    const __bf16* krow = kT + (size_t)(kv0 + Mloc) * CO_ + (hi ? 16 : 0);
    v16bf bfrag = cat8(*(const v8bf*)(krow), *(const v8bf*)(krow + 8));
    v8f sacc = {};
    sacc = __builtin_amdgcn_wmma_f32_16x16x32_bf16(
        false, afrag, false, bfrag, (short)0, sacc, false, false);
#pragma unroll
    for (int j = 0; j < 8; ++j) {
      // |arg| <= 5.78 -> raw v_exp_f32, no denormal fixup needed
      float p = __builtin_amdgcn_exp2f(sacc[j] * TEMP_LOG2E);
      rsum[j] += p;
      int M = hi ? (8 + j) : j;
      Plds[M * KVPAD_ + kv0 + Mloc] = (__bf16)p;
    }
  }

  // rowsum reduce within each 16-lane half (C layout: lanes0-15 rows j,
  // lanes16-31 rows j+8)
#pragma unroll
  for (int j = 0; j < 8; ++j) {
    float v = rsum[j];
    v += __shfl_xor(v, 1, 32);
    v += __shfl_xor(v, 2, 32);
    v += __shfl_xor(v, 4, 32);
    v += __shfl_xor(v, 8, 32);
    rsum[j] = v;
  }
  if (lane == 0) {
#pragma unroll
    for (int j = 0; j < 8; ++j) rs_lds[j] = rsum[j];
  }
  if (lane == 16) {
#pragma unroll
    for (int j = 0; j < 8; ++j) rs_lds[8 + j] = rsum[j];
  }
  __syncthreads();

  float rsden[8];
#pragma unroll
  for (int j = 0; j < 8; ++j) rsden[j] = 1.0f / rs_lds[hi ? 8 + j : j];

  // ---- phase 2: O = (P/rs) . V, 8-wide channel blocking -------------------
  const __bf16* Prow = Plds + Mloc * KVPAD_ + (hi ? 16 : 0);
  const __bf16* vlane = vfrm + (size_t)Mloc * (T_ * HW_);
  for (int ccb = 0; ccb < 4; ++ccb) {
    int c0 = ccb * 128;            // 8 chunks x 16 channels
    v8f oacc[8];
#pragma unroll
    for (int u = 0; u < 8; ++u) {
      v8f z = {};
      oacc[u] = z;
    }
    for (int ks = 0; ks < 25; ++ks) {   // K = 800 (padded) / 32
      int kvb = ks * 32;
      v16bf pa = cat8(*(const v8bf*)(Prow + kvb),
                      *(const v8bf*)(Prow + kvb + 8));
      int  kvs = kvb + (hi ? 16 : 0);
      bool inb = (kvs + 16 <= HW_);
      __builtin_prefetch(vlane + (size_t)c0 * (T_ * HW_) + kvb + 128, 0, 1);
#pragma unroll
      for (int u = 0; u < 8; ++u) {
        const __bf16* vcol = vlane + (size_t)(c0 + u * 16) * (T_ * HW_);
        v16bf vb;
        if (inb) {
          vb = cat8(*(const v8bf*)(vcol + kvs), *(const v8bf*)(vcol + kvs + 8));
        } else {
          v16bf z = {};
          vb = z;                       // kv in [784,800)
        }
        oacc[u] = __builtin_amdgcn_wmma_f32_16x16x32_bf16(
            false, pa, false, vb, (short)0, oacc[u], false, false);
      }
    }
#pragma unroll
    for (int u = 0; u < 8; ++u) {
      float* optr = obase + (size_t)(c0 + u * 16 + Mloc) * (TOUT_ * HW_);
      float4 s0 = {oacc[u][0] * rsden[0], oacc[u][1] * rsden[1],
                   oacc[u][2] * rsden[2], oacc[u][3] * rsden[3]};
      float4 s1 = {oacc[u][4] * rsden[4], oacc[u][5] * rsden[5],
                   oacc[u][6] * rsden[6], oacc[u][7] * rsden[7]};
      *(float4*)(optr)     = s0;       // p = m0 + (hi?8:0) + 0..3
      *(float4*)(optr + 4) = s1;       // p = m0 + (hi?8:0) + 4..7
    }
  }
}

// ---------------------------------------------------------------------------
// Kernel 3: gating att = sigmoid(Watt . (Wx@x * Wn@O)), scale O in place in
// d_out, and copy x into the center temporal slices.
// ---------------------------------------------------------------------------
__global__ __launch_bounds__(256) void gate_kernel(
    const float* __restrict__ x, const float* __restrict__ Wx,
    const float* __restrict__ Wn, const float* __restrict__ Watt,
    float* __restrict__ out) {
  int idx = blockIdx.x * blockDim.x + threadIdx.x;
  if (idx >= B_ * T_ * 2 * HW_) return;
  int p    = idx % HW_;
  int rest = idx / HW_;
  int n    = rest & 1;
  int t    = (rest >> 1) % T_;
  int b    = rest / (2 * T_);
  int tout = t * 3 + (n ? 2 : 0);

  const float* xb   = x + ((size_t)b * C_) * (T_ * HW_) + (size_t)t * HW_ + p;
  float*       ob   = out + ((size_t)b * C_) * (TOUT_ * HW_) + (size_t)tout * HW_ + p;
  float*       ocen = out + ((size_t)b * C_) * (TOUT_ * HW_) + (size_t)(t * 3 + 1) * HW_ + p;

  float xatt[CO_], natt[CO_];
#pragma unroll
  for (int o = 0; o < CO_; ++o) { xatt[o] = 0.f; natt[o] = 0.f; }

  for (int c = 0; c < C_; ++c) {
    float xv = xb[(size_t)c * (T_ * HW_)];
    float nv = ob[(size_t)c * (TOUT_ * HW_)];
    if (n == 0) ocen[(size_t)c * (TOUT_ * HW_)] = xv;  // center slice = x
#pragma unroll
    for (int o = 0; o < CO_; ++o) {
      xatt[o] += Wx[o * C_ + c] * xv;
      natt[o] += Wn[o * C_ + c] * nv;
    }
  }
  float s = 0.f;
#pragma unroll
  for (int o = 0; o < CO_; ++o) s += Watt[o] * xatt[o] * natt[o];
  float att = 1.0f / (1.0f + __expf(-s));

  for (int c = 0; c < C_; ++c) ob[(size_t)c * (TOUT_ * HW_)] *= att;
}

// ---------------------------------------------------------------------------
extern "C" void kernel_launch(void* const* d_in, const int* in_sizes, int n_in,
                              void* d_out, int out_size, void* d_ws, size_t ws_size,
                              hipStream_t stream) {
  const float* x    = (const float*)d_in[0];
  const float* Ws   = (const float*)d_in[1];
  const float* Wx   = (const float*)d_in[2];
  const float* Wn   = (const float*)d_in[3];
  const float* Watt = (const float*)d_in[4];
  float* out = (float*)d_out;

  const size_t xbf_elems = (size_t)B_ * C_ * T_ * HW_;     // 25,690,112
  __bf16* xbf = (__bf16*)d_ws;                             // 51.4 MB
  __bf16* xn  = (__bf16*)((char*)d_ws + xbf_elems * 2);    // 3.2 MB, (b,t,hw,co)

  cvt_kernel<<<(int)(xbf_elems / (256 * 8)), 256, 0, stream>>>(x, xbf);
  semnorm_kernel<<<B_ * T_, 256, 0, stream>>>(x, Ws, xn);
  attn_kernel<<<B_ * T_ * 2 * 49, 32, 0, stream>>>(xbf, xn, out);
  int tot = B_ * T_ * 2 * HW_;
  gate_kernel<<<(tot + 255) / 256, 256, 0, stream>>>(x, Wx, Wn, Watt, out);
}
